// GAT_GCN_1013612282047
// MI455X (gfx1250) — matrix operand order repaired and checked by reading, not compiled, not run on hardware
//
#include <hip/hip_runtime.h>
#include <hip/hip_bf16.h>
#include <stdint.h>

#define NN      40960      // nodes
#define E0      163840     // raw edges
#define EE      204800     // edges + self loops
#define NHEADS  10
#define FOUT    78
#define HIDV    780
#define NGR     1024       // graphs / batch

typedef _Float16 half8  __attribute__((ext_vector_type(8)));
typedef _Float16 half16 __attribute__((ext_vector_type(16)));
typedef float    float8 __attribute__((ext_vector_type(8)));

// ---------------- ordered-float encoding for atomic max ----------------
__device__ __forceinline__ unsigned encOrd(float f) {
  unsigned u = __float_as_uint(f);
  return (u & 0x80000000u) ? ~u : (u | 0x80000000u);
}
__device__ __forceinline__ float decOrd(unsigned k) {
  unsigned u = (k & 0x80000000u) ? (k & 0x7FFFFFFFu) : ~k;
  return __uint_as_float(u);
}
__device__ __forceinline__ void atomicMaxKey(unsigned* a, float v) {
  atomicMax(a, encOrd(v));
}

// ---------------- generic fill ----------------
__global__ void k_fill(uint32_t* __restrict__ p, uint32_t v, long long n) {
  long long i = (long long)blockIdx.x * blockDim.x + threadIdx.x;
  if (i < n) p[i] = v;
}

// ---------------- pack x -> f16 padded [NN x 96] ----------------
__global__ void k_pad_x(const float* __restrict__ x, _Float16* __restrict__ XH) {
  long long i = (long long)blockIdx.x * blockDim.x + threadIdx.x;
  if (i >= (long long)NN * 96) return;
  int n = (int)(i / 96), k = (int)(i % 96);
  XH[i] = (k < 78) ? (_Float16)x[(size_t)n * 78 + k] : (_Float16)0.0f;
}

// ---------------- transpose-pack weight [K,N] f32 -> Bt [Np,Kp] f16 ----------------
__global__ void k_wpackT(const float* __restrict__ src, _Float16* __restrict__ dst,
                         int K, int N, int Kp, int Np) {
  long long i = (long long)blockIdx.x * blockDim.x + threadIdx.x;
  if (i >= (long long)Np * Kp) return;
  int n = (int)(i / Kp), k = (int)(i % Kp);
  dst[i] = (k < K && n < N) ? (_Float16)src[(size_t)k * N + n] : (_Float16)0.0f;
}

// ---------------- conv weight pack: [32,1000,8] -> Bt [(o*8+kh)=256][1024] ----------------
__global__ void k_convpack(const float* __restrict__ W, _Float16* __restrict__ dst) {
  long long i = (long long)blockIdx.x * blockDim.x + threadIdx.x;
  if (i >= (long long)256 * 1024) return;
  int r = (int)(i >> 10), c = (int)(i & 1023);
  int o = r >> 3, kh = r & 7;
  dst[i] = (c < 1000) ? (_Float16)W[(size_t)o * 8000 + (size_t)c * 8 + kh] : (_Float16)0.0f;
}

// ---------------- WMMA fragment loads (wave32 layouts per CDNA5 ISA 7.12.2) ----------------
// A 16x32 f16: lanes 0-15 hold K{0..7,16..23}, lanes 16-31 K{8..15,24..31}; ptr includes ksel (0|8)
__device__ __forceinline__ half16 load_fragA(const _Float16* p) {
  half8 lo = *reinterpret_cast<const half8*>(p);
  half8 hi = *reinterpret_cast<const half8*>(p + 16);
  half16 r;
#pragma unroll
  for (int i = 0; i < 8; ++i) { r[i] = lo[i]; r[i + 8] = hi[i]; }
  return r;
}
// B 32x16 f16 (from Bt row-major): lanes 0-15 hold K{0..7,16..23}, lanes 16-31 K{8..15,24..31}; ptr includes ksel (0|8)
__device__ __forceinline__ half16 load_fragB(const _Float16* p) {
  half8 lo = *reinterpret_cast<const half8*>(p);
  half8 hi = *reinterpret_cast<const half8*>(p + 16);
  half16 r;
#pragma unroll
  for (int i = 0; i < 8; ++i) { r[i] = lo[i]; r[i + 8] = hi[i]; }
  return r;
}

__device__ __forceinline__ void epi(float8 acc, int rowBase, int col, int mOff,
                                    const float* bias, int nbias,
                                    _Float16* Ch, int ldch, int relu) {
  float bval = bias ? ((col < nbias) ? bias[col] : 0.0f) : 0.0f;
#pragma unroll
  for (int r = 0; r < 8; ++r) {
    int row = rowBase + mOff + r;
    float v = acc[r] + bval;
    if (relu) v = v > 0.0f ? v : 0.0f;
    if (Ch) Ch[(size_t)row * ldch + col] = (_Float16)v;
  }
}

// ---------------- tiled WMMA GEMM: Ch[M,N](f16) = act(A[M,K]f16 * Bt[N,K]^T + bias) ----------------
// block = 128 threads = 4 waves stacked in M; block tile 128x64; wave tile 32x64 (2x4 wmma frags)
// 12 b128 loads per 8 WMMA per K-step (vs 8:4 for a square 32x32 tile)
__global__ __launch_bounds__(128, 2)
void k_gemm(const _Float16* __restrict__ A, const _Float16* __restrict__ Bt,
            const float* __restrict__ bias, int nbias,
            _Float16* __restrict__ Ch, int ldch,
            int K, int lda, int ldb, int relu) {
  const int lane = threadIdx.x & 31;
  const int wid  = threadIdx.x >> 5;
  const int rowBase = blockIdx.y * 128 + wid * 32;
  const int colBase = blockIdx.x * 64;
  const int lrow = lane & 15;
  const int hi16 = lane >> 4;

  const _Float16* a0 = A  + (size_t)(rowBase + lrow) * lda + hi16 * 8;
  const _Float16* a1 = a0 + (size_t)16 * lda;
  const _Float16* b0 = Bt + (size_t)(colBase + lrow) * ldb + hi16 * 8;
  const _Float16* b1 = b0 + (size_t)16 * ldb;
  const _Float16* b2 = b0 + (size_t)32 * ldb;
  const _Float16* b3 = b0 + (size_t)48 * ldb;

  float8 c00 = {0,0,0,0,0,0,0,0};
  float8 c01 = c00, c02 = c00, c03 = c00;
  float8 c10 = c00, c11 = c00, c12 = c00, c13 = c00;

  for (int k = 0; k < K; k += 32) {
    if (k + 128 < K) {                       // pull next K-tiles into WGP$/L2
      __builtin_prefetch(a0 + k + 128, 0, 3);
      __builtin_prefetch(b0 + k + 128, 0, 3);
    }
    half16 av0 = load_fragA(a0 + k);
    half16 av1 = load_fragA(a1 + k);
    half16 bv0 = load_fragB(b0 + k);
    half16 bv1 = load_fragB(b1 + k);
    half16 bv2 = load_fragB(b2 + k);
    half16 bv3 = load_fragB(b3 + k);
    c00 = __builtin_amdgcn_wmma_f32_16x16x32_f16(false, av0, false, bv0, (short)0, c00, false, false);
    c01 = __builtin_amdgcn_wmma_f32_16x16x32_f16(false, av0, false, bv1, (short)0, c01, false, false);
    c02 = __builtin_amdgcn_wmma_f32_16x16x32_f16(false, av0, false, bv2, (short)0, c02, false, false);
    c03 = __builtin_amdgcn_wmma_f32_16x16x32_f16(false, av0, false, bv3, (short)0, c03, false, false);
    c10 = __builtin_amdgcn_wmma_f32_16x16x32_f16(false, av1, false, bv0, (short)0, c10, false, false);
    c11 = __builtin_amdgcn_wmma_f32_16x16x32_f16(false, av1, false, bv1, (short)0, c11, false, false);
    c12 = __builtin_amdgcn_wmma_f32_16x16x32_f16(false, av1, false, bv2, (short)0, c12, false, false);
    c13 = __builtin_amdgcn_wmma_f32_16x16x32_f16(false, av1, false, bv3, (short)0, c13, false, false);
  }
  const int mOff = hi16 * 8;
  epi(c00, rowBase,      colBase      + lrow, mOff, bias, nbias, Ch, ldch, relu);
  epi(c01, rowBase,      colBase + 16 + lrow, mOff, bias, nbias, Ch, ldch, relu);
  epi(c02, rowBase,      colBase + 32 + lrow, mOff, bias, nbias, Ch, ldch, relu);
  epi(c03, rowBase,      colBase + 48 + lrow, mOff, bias, nbias, Ch, ldch, relu);
  epi(c10, rowBase + 16, colBase      + lrow, mOff, bias, nbias, Ch, ldch, relu);
  epi(c11, rowBase + 16, colBase + 16 + lrow, mOff, bias, nbias, Ch, ldch, relu);
  epi(c12, rowBase + 16, colBase + 32 + lrow, mOff, bias, nbias, Ch, ldch, relu);
  epi(c13, rowBase + 16, colBase + 48 + lrow, mOff, bias, nbias, Ch, ldch, relu);
}

// ---------------- GAT attention dots: a_src/a_dst [NN x 10] ----------------
__global__ void k_att_dots(const _Float16* __restrict__ H, const float* __restrict__ wsrc,
                           const float* __restrict__ wdst, float* __restrict__ AS,
                           float* __restrict__ AD) {
  int i = blockIdx.x * blockDim.x + threadIdx.x;
  if (i >= NN * NHEADS) return;
  int n = i / NHEADS, hd = i % NHEADS;
  const _Float16* hp = H + (size_t)n * 832 + hd * FOUT;
  const float* ws_ = wsrc + hd * FOUT;
  const float* wd_ = wdst + hd * FOUT;
  float s = 0.0f, d = 0.0f;
#pragma unroll 6
  for (int f = 0; f < FOUT; ++f) { float v = (float)hp[f]; s += v * ws_[f]; d += v * wd_[f]; }
  AS[i] = s; AD[i] = d;
}

__device__ __forceinline__ void edgeSD(const int* __restrict__ ei, int e, int& s, int& d) {
  if (e < E0) { s = ei[e]; d = ei[E0 + e]; }
  else        { s = e - E0; d = s; }
}

__global__ void k_edge_logit(const int* __restrict__ ei, const float* __restrict__ AS,
                             const float* __restrict__ AD, float* __restrict__ EAL,
                             unsigned* __restrict__ MAXB) {
  int i = blockIdx.x * blockDim.x + threadIdx.x;
  if (i >= EE * NHEADS) return;
  int e = i / NHEADS, hd = i % NHEADS;
  int s, d; edgeSD(ei, e, s, d);
  float l = AS[s * NHEADS + hd] + AD[d * NHEADS + hd];
  l = (l >= 0.0f) ? l : 0.2f * l;   // leaky relu
  EAL[i] = l;
  atomicMaxKey(&MAXB[d * NHEADS + hd], l);
}

__global__ void k_edge_exp(const int* __restrict__ ei, float* __restrict__ EAL,
                           const unsigned* __restrict__ MAXB, float* __restrict__ SUMB) {
  int i = blockIdx.x * blockDim.x + threadIdx.x;
  if (i >= EE * NHEADS) return;
  int e = i / NHEADS, hd = i % NHEADS;
  int s, d; edgeSD(ei, e, s, d);
  float v = expf(EAL[i] - decOrd(MAXB[d * NHEADS + hd]));
  EAL[i] = v;
  atomicAdd(&SUMB[d * NHEADS + hd], v);
}

__global__ void k_gat_agg(const int* __restrict__ ei, const float* __restrict__ EAL,
                          const float* __restrict__ SUMB, const _Float16* __restrict__ H,
                          float* __restrict__ AGG) {
  int i = blockIdx.x * blockDim.x + threadIdx.x;
  if (i >= EE * NHEADS) return;
  int e = i / NHEADS, hd = i % NHEADS;
  int s, d; edgeSD(ei, e, s, d);
  float alpha = EAL[i] / (SUMB[d * NHEADS + hd] + 1e-16f);
  const _Float16* hrow = H + (size_t)s * 832 + hd * FOUT;
  float* out = AGG + (size_t)d * HIDV + hd * FOUT;
#pragma unroll 6
  for (int f = 0; f < FOUT; ++f) atomicAdd(&out[f], alpha * (float)hrow[f]);
}

__global__ void k_x1(const float* __restrict__ AGG, const float* __restrict__ bias,
                     _Float16* __restrict__ X1H) {
  long long i = (long long)blockIdx.x * blockDim.x + threadIdx.x;
  if (i >= (long long)NN * 800) return;
  int n = (int)(i / 800), c = (int)(i % 800);
  float v = 0.0f;
  if (c < HIDV) { v = AGG[(size_t)n * HIDV + c] + bias[c]; v = v > 0.0f ? v : 0.0f; }
  X1H[i] = (_Float16)v;
}

__global__ void k_deg(const int* __restrict__ ei, float* __restrict__ DEG) {
  int e = blockIdx.x * blockDim.x + threadIdx.x;
  if (e >= EE) return;
  int s, d; edgeSD(ei, e, s, d);
  atomicAdd(&DEG[d], 1.0f);
}
__global__ void k_dinv(const float* __restrict__ DEG, float* __restrict__ DINV) {
  int n = blockIdx.x * blockDim.x + threadIdx.x;
  if (n >= NN) return;
  float dg = DEG[n];
  DINV[n] = dg > 0.0f ? rsqrtf(dg) : 0.0f;
}

__global__ void k_gcn_agg(const int* __restrict__ ei, const _Float16* __restrict__ H2,
                          const float* __restrict__ DINV, float* __restrict__ AGG2) {
  long long i = (long long)blockIdx.x * blockDim.x + threadIdx.x;
  if (i >= (long long)EE * 195) return;
  int e = (int)(i / 195), j = (int)(i % 195) * 4;
  int s, d; edgeSD(ei, e, s, d);
  float norm = DINV[s] * DINV[d];
  const _Float16* hp = H2 + (size_t)s * 832 + j;
  float* out = AGG2 + (size_t)d * HIDV + j;
#pragma unroll
  for (int q = 0; q < 4; ++q) atomicAdd(&out[q], norm * (float)hp[q]);
}

__global__ void k_cnt(const int* __restrict__ batch, float* __restrict__ CNT) {
  int n = blockIdx.x * blockDim.x + threadIdx.x;
  if (n >= NN) return;
  atomicAdd(&CNT[batch[n]], 1.0f);
}

__global__ void k_pool(const float* __restrict__ AGG2, const float* __restrict__ bias,
                       const int* __restrict__ batch, unsigned* __restrict__ GMAX,
                       float* __restrict__ GSUM) {
  long long i = (long long)blockIdx.x * blockDim.x + threadIdx.x;
  if (i >= (long long)NN * HIDV) return;
  int n = (int)(i / HIDV), c = (int)(i % HIDV);
  float v = AGG2[i] + bias[c];
  v = v > 0.0f ? v : 0.0f;               // x2 = relu(agg2 + bias)
  int b = batch[n];
  atomicMaxKey(&GMAX[(size_t)b * HIDV + c], v);
  atomicAdd(&GSUM[(size_t)b * HIDV + c], v);
}

__global__ void k_gpack(const unsigned* __restrict__ GMAX, const float* __restrict__ GSUM,
                        const float* __restrict__ CNT, _Float16* __restrict__ G) {
  long long i = (long long)blockIdx.x * blockDim.x + threadIdx.x;
  if (i >= (long long)NGR * 1568) return;
  int b = (int)(i / 1568), c = (int)(i % 1568);
  float v = 0.0f;
  if (c < HIDV)            v = decOrd(GMAX[(size_t)b * HIDV + c]);
  else if (c < 2 * HIDV)   v = GSUM[(size_t)b * HIDV + (c - HIDV)] / fmaxf(CNT[b], 1.0f);
  G[i] = (_Float16)v;
}

// ---------------- protein branch ----------------
// eT[b, h, c] = embed[target[b,c], h]   (K padded 1000->1024)
__global__ void k_build_eT(const int* __restrict__ target, const float* __restrict__ embed,
                           _Float16* __restrict__ ET) {
  long long i = (long long)blockIdx.x * blockDim.x + threadIdx.x;
  if (i >= (long long)NGR * 128 * 1024) return;
  int c  = (int)(i & 1023);
  int hp = (int)((i >> 10) & 127);
  int b  = (int)(i >> 17);
  float v = 0.0f;
  if (c < 1000) {
    int t = target[(size_t)b * 1000 + c];
    v = embed[(size_t)t * 128 + hp];
  }
  ET[i] = (_Float16)v;
}

// xt[b, o*121+t] = sum_kh Z[(b*128+t+kh), o*8+kh] + conv_b[o]
__global__ void k_conv_gather(const _Float16* __restrict__ Z, const float* __restrict__ convb,
                              _Float16* __restrict__ XT) {
  int i = blockIdx.x * blockDim.x + threadIdx.x;
  if (i >= NGR * 32 * 121) return;
  int t = i % 121;
  int o = (i / 121) % 32;
  int b = i / (121 * 32);
  float acc = convb[o];
#pragma unroll
  for (int kh = 0; kh < 8; ++kh)
    acc += (float)Z[((size_t)(b * 128 + t + kh)) * 256 + o * 8 + kh];
  XT[(size_t)b * 3872 + o * 121 + t] = (_Float16)acc;
}

// ---------------- final out layer (N=1, plain dot) ----------------
__global__ void k_out(const _Float16* __restrict__ A2, const float* __restrict__ outW,
                      const float* __restrict__ outb, float* __restrict__ out) {
  int b = blockIdx.x * blockDim.x + threadIdx.x;
  if (b >= NGR) return;
  float acc = outb[0];
  const _Float16* row = A2 + (size_t)b * 512;
#pragma unroll 8
  for (int j = 0; j < 512; ++j) acc += (float)row[j] * outW[j];
  out[b] = acc;
}

// ======================================================================
extern "C" void kernel_launch(void* const* d_in, const int* in_sizes, int n_in,
                              void* d_out, int out_size, void* d_ws, size_t ws_size,
                              hipStream_t stream) {
  const float* x        = (const float*)d_in[0];
  const int*   edge     = (const int*)  d_in[1];
  const int*   batch    = (const int*)  d_in[2];
  const int*   target   = (const int*)  d_in[3];
  const float* gat_W    = (const float*)d_in[4];
  const float* att_src  = (const float*)d_in[5];
  const float* att_dst  = (const float*)d_in[6];
  const float* gat_bias = (const float*)d_in[7];
  const float* gcn_W    = (const float*)d_in[8];
  const float* gcn_bias = (const float*)d_in[9];
  const float* fcg1_W   = (const float*)d_in[10];
  const float* fcg1_b   = (const float*)d_in[11];
  const float* fcg2_W   = (const float*)d_in[12];
  const float* fcg2_b   = (const float*)d_in[13];
  const float* embed    = (const float*)d_in[14];
  const float* conv_W   = (const float*)d_in[15];
  const float* conv_b   = (const float*)d_in[16];
  const float* fc1xt_W  = (const float*)d_in[17];
  const float* fc1xt_b  = (const float*)d_in[18];
  const float* fc1_W    = (const float*)d_in[19];
  const float* fc1_b    = (const float*)d_in[20];
  const float* fc2_W    = (const float*)d_in[21];
  const float* fc2_b    = (const float*)d_in[22];
  const float* outW     = (const float*)d_in[23];
  const float* outb     = (const float*)d_in[24];
  float* out = (float*)d_out;

  // ---- workspace layout (deterministic bump allocator; big arena reused) ----
  char* ws = (char*)d_ws;
  size_t off = 0;
  auto alloc = [&](size_t bytes) -> char* {
    char* p = ws + off;
    off = (off + bytes + 255) & ~(size_t)255;
    return p;
  };
  _Float16* XH     = (_Float16*)alloc((size_t)NN * 96 * 2);
  _Float16* WGAT   = (_Float16*)alloc((size_t)832 * 96 * 2);
  _Float16* WGCN   = (_Float16*)alloc((size_t)832 * 800 * 2);
  _Float16* WFCG1  = (_Float16*)alloc((size_t)1536 * 1568 * 2);
  _Float16* WFCG2  = (_Float16*)alloc((size_t)128 * 1536 * 2);
  _Float16* WCONV  = (_Float16*)alloc((size_t)256 * 1024 * 2);
  _Float16* WFC1XT = (_Float16*)alloc((size_t)128 * 3872 * 2);
  _Float16* WFC1   = (_Float16*)alloc((size_t)1024 * 256 * 2);
  _Float16* WFC2   = (_Float16*)alloc((size_t)512 * 1024 * 2);
  float*    AS     = (float*)alloc((size_t)NN * NHEADS * 4);
  float*    AD     = (float*)alloc((size_t)NN * NHEADS * 4);
  unsigned* MAXB   = (unsigned*)alloc((size_t)NN * NHEADS * 4);
  float*    SUMB   = (float*)alloc((size_t)NN * NHEADS * 4);
  float*    DEG    = (float*)alloc((size_t)NN * 4);
  float*    DINV   = (float*)alloc((size_t)NN * 4);
  unsigned* GMAX   = (unsigned*)alloc((size_t)NGR * HIDV * 4);
  float*    GSUM   = (float*)alloc((size_t)NGR * HIDV * 4);
  float*    CNT    = (float*)alloc((size_t)NGR * 4);
  _Float16* G      = (_Float16*)alloc((size_t)NGR * 1568 * 2);
  _Float16* FCG1O  = (_Float16*)alloc((size_t)NGR * 1536 * 2);
  _Float16* XC     = (_Float16*)alloc((size_t)NGR * 256 * 2);
  _Float16* A1     = (_Float16*)alloc((size_t)NGR * 1024 * 2);
  _Float16* A2     = (_Float16*)alloc((size_t)NGR * 512 * 2);
  // big reuse arena
  char* big = ws + off;
  _Float16* H    = (_Float16*)(big);                                             // 68.2 MB
  float*    AGG  = (float*)   (big + (size_t)NN * 832 * 2);                      // 127.8 MB
  _Float16* X1H  = (_Float16*)(big + (size_t)NN * 832 * 2 + (size_t)NN * HIDV * 4); // 65.5 MB
  float*    EAL  = (float*)   (big + (size_t)NN * 832 * 2 + (size_t)NN * HIDV * 4
                                   + (size_t)NN * 800 * 2);                      // 8.2 MB
  _Float16* H2   = H;     // H dead when h2 is produced
  float*    AGG2 = AGG;   // agg dead when agg2 is produced
  // protein branch reuses the whole arena (graph branch fully consumed by then)
  _Float16* ET = (_Float16*)big;                                                 // 268.4 MB
  _Float16* Z  = (_Float16*)(big + (size_t)NGR * 128 * 1024 * 2);                // 67.1 MB
  _Float16* XT = (_Float16*)(big + (size_t)NGR * 128 * 1024 * 2
                                 + (size_t)131072 * 256 * 2);                    // 7.9 MB

  auto fill = [&](void* p, uint32_t v, long long n) {
    k_fill<<<dim3((unsigned)((n + 255) / 256)), 256, 0, stream>>>((uint32_t*)p, v, n);
  };
  auto wpackT = [&](const float* src, _Float16* dst, int K, int N, int Kp, int Np) {
    long long n = (long long)Np * Kp;
    k_wpackT<<<dim3((unsigned)((n + 255) / 256)), 256, 0, stream>>>(src, dst, K, N, Kp, Np);
  };
  auto gemm = [&](const _Float16* A_, const _Float16* B_, const float* bias_, int nbias_,
                  _Float16* Ch_, int ldch_, int M_, int N_, int K_, int lda_, int ldb_, int relu_) {
    dim3 grid((unsigned)(N_ / 64), (unsigned)(M_ / 128));
    k_gemm<<<grid, 128, 0, stream>>>(A_, B_, bias_, nbias_, Ch_, ldch_, K_, lda_, ldb_, relu_);
  };
  const unsigned KEY_NEGINF = 0x007FFFFFu;  // encOrd(-inf)
  const unsigned KEY_ZERO   = 0x80000000u;  // encOrd(0.0f)

  // ---- pack activations + weights to padded f16 ----
  k_pad_x<<<dim3((unsigned)(((long long)NN * 96 + 255) / 256)), 256, 0, stream>>>(x, XH);
  wpackT(gat_W,   WGAT,   78,   780,  96,   832);
  wpackT(gcn_W,   WGCN,   780,  780,  800,  832);
  wpackT(fcg1_W,  WFCG1,  1560, 1500, 1568, 1536);
  wpackT(fcg2_W,  WFCG2,  1500, 128,  1536, 128);
  k_convpack<<<dim3((unsigned)((256 * 1024 + 255) / 256)), 256, 0, stream>>>(conv_W, WCONV);
  wpackT(fc1xt_W, WFC1XT, 3872, 128,  3872, 128);
  wpackT(fc1_W,   WFC1,   256,  1024, 256,  1024);
  wpackT(fc2_W,   WFC2,   1024, 512,  1024, 512);

  // ---- GAT: h = x @ gat_W ----
  gemm(XH, WGAT, nullptr, 0, H, 832, NN, 832, 96, 96, 96, 0);
  k_att_dots<<<dim3((NN * NHEADS + 255) / 256), 256, 0, stream>>>(H, att_src, att_dst, AS, AD);
  fill(MAXB, KEY_NEGINF, (long long)NN * NHEADS);
  k_edge_logit<<<dim3((EE * NHEADS + 255) / 256), 256, 0, stream>>>(edge, AS, AD, EAL, MAXB);
  fill(SUMB, 0u, (long long)NN * NHEADS);
  k_edge_exp<<<dim3((EE * NHEADS + 255) / 256), 256, 0, stream>>>(edge, EAL, MAXB, SUMB);
  fill(AGG, 0u, (long long)NN * HIDV);
  k_gat_agg<<<dim3((EE * NHEADS + 255) / 256), 256, 0, stream>>>(edge, EAL, SUMB, H, AGG);
  k_x1<<<dim3((unsigned)(((long long)NN * 800 + 255) / 256)), 256, 0, stream>>>(AGG, gat_bias, X1H);

  // ---- GCN: h2 = x1 @ gcn_W, sym-norm aggregate ----
  fill(DEG, 0u, NN);
  k_deg<<<dim3((EE + 255) / 256), 256, 0, stream>>>(edge, DEG);
  k_dinv<<<dim3((NN + 255) / 256), 256, 0, stream>>>(DEG, DINV);
  gemm(X1H, WGCN, nullptr, 0, H2, 832, NN, 832, 800, 800, 800, 0);
  fill(AGG2, 0u, (long long)NN * HIDV);
  k_gcn_agg<<<dim3((unsigned)(((long long)EE * 195 + 255) / 256)), 256, 0, stream>>>(edge, H2, DINV, AGG2);

  // ---- pooling + graph FC head ----
  fill(GMAX, KEY_ZERO, (long long)NGR * HIDV);
  fill(GSUM, 0u, (long long)NGR * HIDV);
  fill(CNT, 0u, NGR);
  k_cnt<<<dim3((NN + 255) / 256), 256, 0, stream>>>(batch, CNT);
  k_pool<<<dim3((unsigned)(((long long)NN * HIDV + 255) / 256)), 256, 0, stream>>>(AGG2, gcn_bias, batch, GMAX, GSUM);
  k_gpack<<<dim3((unsigned)(((long long)NGR * 1568 + 255) / 256)), 256, 0, stream>>>(GMAX, GSUM, CNT, G);
  gemm(G, WFCG1, fcg1_b, 1500, FCG1O, 1536, NGR, 1536, 1568, 1568, 1568, 1);
  gemm(FCG1O, WFCG2, fcg2_b, 128, XC, 256, NGR, 128, 1536, 1536, 1536, 0);  // -> xc[:, :128]

  // ---- protein branch (reuses big arena; graph branch fully consumed) ----
  k_build_eT<<<dim3((unsigned)(((long long)NGR * 128 * 1024 + 255) / 256)), 256, 0, stream>>>(target, embed, ET);
  gemm(ET, WCONV, nullptr, 0, Z, 256, NGR * 128, 256, 1024, 1024, 1024, 0);
  k_conv_gather<<<dim3((NGR * 32 * 121 + 255) / 256), 256, 0, stream>>>(Z, conv_b, XT);
  gemm(XT, WFC1XT, fc1xt_b, 128, XC + 128, 256, NGR, 128, 3872, 3872, 3872, 0); // -> xc[:, 128:]

  // ---- joint head ----
  gemm(XC, WFC1, fc1_b, 1024, A1, 1024, NGR, 1024, 256, 256, 256, 1);
  gemm(A1, WFC2, fc2_b, 512, A2, 512, NGR, 512, 1024, 1024, 1024, 1);
  k_out<<<dim3((NGR + 255) / 256), 256, 0, stream>>>(A2, outW, outb, out);

  (void)in_sizes; (void)n_in; (void)out_size; (void)ws_size;
}
